// ComplexMultiHeadAttention_13374528160387
// MI455X (gfx1250) — matrix operand order repaired and checked by reading, not compiled, not run on hardware
//
#include <hip/hip_runtime.h>

// Problem constants (B,S,D,H from reference)
constexpr int CB = 8, CS = 1024, CD = 512, CH = 8, CDH = 64;

typedef __attribute__((ext_vector_type(16))) __bf16 v16bf;
typedef __attribute__((ext_vector_type(8)))  __bf16 v8bf;
typedef __attribute__((ext_vector_type(8)))  float  v8f;
typedef __attribute__((ext_vector_type(4)))  unsigned int uint4v;
typedef __attribute__((ext_vector_type(8)))  int int8v;
typedef __attribute__((ext_vector_type(4)))  int int4v;

__device__ __forceinline__ __bf16 f2bf(float f) {
  unsigned u = __builtin_bit_cast(unsigned, f);
  unsigned r = u + 0x7FFFu + ((u >> 16) & 1u);   // round-to-nearest-even
  unsigned short h = (unsigned short)(r >> 16);
  return __builtin_bit_cast(__bf16, h);
}

// bf16 WMMA: NEG/NEG_HI must be 0 for bf16 (only CNeg exists, unused here)
__device__ __forceinline__ v8f wmma_bf16(v16bf a, v16bf b, v8f c) {
  return __builtin_amdgcn_wmma_f32_16x16x32_bf16(false, a, false, b, (short)0, c,
                                                 false, false);
}

__device__ __forceinline__ v8bf ld8(const __bf16* p)  { return *(const v8bf*)p;  }
__device__ __forceinline__ v16bf ld16(const __bf16* p){ return *(const v16bf*)p; }
__device__ __forceinline__ v16bf concat8(v8bf lo, v8bf hi) {
  return __builtin_shufflevector(lo, hi, 0,1,2,3,4,5,6,7,8,9,10,11,12,13,14,15);
}

// --- Tensor Data Mover: 2-D tile global->LDS (ISA cdna5 ch.8, D# groups 0/1) --
__device__ __forceinline__ unsigned ldsOff(const void* p) {
  return (unsigned)(unsigned long long)p;   // LDS aperture: offset in addr[31:0]
}
__device__ __forceinline__ void tdm_load_2d(unsigned lds, const void* gptr,
                                            unsigned tensor_d0, unsigned tensor_d1,
                                            unsigned pitch_d0,
                                            unsigned tile_d0, unsigned tile_d1) {
  unsigned long long ga = (unsigned long long)gptr;
  uint4v g0;
  g0[0] = 1u;                                            // count=1 (valid user D#)
  g0[1] = lds;                                           // lds_addr  [63:32]
  g0[2] = (unsigned)(ga & 0xFFFFFFFFu);                  // global_addr[31:0]
  g0[3] = (unsigned)((ga >> 32) & 0x1FFFFFFu) | (2u << 30); // addr[56:32] | type=2
  int8v g1;
  g1[0] = (int)(1u << 16);                               // data_size=1 (2B), mask=0
  g1[1] = (int)((tensor_d0 & 0xFFFFu) << 16);            // tensor_dim0 lo -> bits 63:48
  g1[2] = (int)(((tensor_d0 >> 16) & 0xFFFFu) |          // tensor_dim0 hi
                ((tensor_d1 & 0xFFFFu) << 16));          // tensor_dim1 lo
  g1[3] = (int)(((tensor_d1 >> 16) & 0xFFFFu) |          // tensor_dim1 hi
                ((tile_d0 & 0xFFFFu) << 16));            // tile_dim0 -> bits 127:112
  g1[4] = (int)(tile_d1 & 0xFFFFu);                      // tile_dim1; tile_dim2=0
  g1[5] = (int)pitch_d0;                                 // tensor_dim0_stride lo
  g1[6] = 0;                                             // stride hi / dim1_stride lo
  g1[7] = 0;
  int4v z4 = {0, 0, 0, 0};
  int8v z8 = {0, 0, 0, 0, 0, 0, 0, 0};
  // clang-23 (amdgpu-toolchain) 6-arg form: g0, g1, g2, g3, extra group, cpol
  __builtin_amdgcn_tensor_load_to_lds(g0, g1, z4, z4, z8, 0);
}

// ---------------------------------------------------------------------------
// Prep kernels: one-time fp32 -> bf16 conversions (kept out of GEMM hot loops)
// ---------------------------------------------------------------------------
__global__ void __launch_bounds__(256)
conv_x(const float2* __restrict__ X, __bf16* __restrict__ xr, __bf16* __restrict__ xi) {
  int i = blockIdx.x * 256 + threadIdx.x;     // M*K elements
  float2 c = X[i];
  xr[i] = f2bf(c.x);
  xi[i] = f2bf(c.y);
}

__global__ void __launch_bounds__(256)
conv_w(const float* __restrict__ Wr, const float* __restrict__ Wi,
       __bf16* __restrict__ br, __bf16* __restrict__ bi, __bf16* __restrict__ bn) {
  int i = blockIdx.x * 256 + threadIdx.x;     // 512*512 elements
  float wr = Wr[i], wi = Wi[i];
  br[i] = f2bf(wr);
  bi[i] = f2bf(wi);
  bn[i] = f2bf(-wi);                          // bf16 NEG bits must be 0 -> pre-negate
}

// ---------------------------------------------------------------------------
// Complex GEMM on pre-converted bf16 operands:  Y = X @ W^T (complex).
// Xr/Xi: [M=8192, K=512] bf16 planes; Br/Bi/Bn(-Wi): [N=512, K=512] bf16.
// One wave computes a 16x64 complex tile (A fragments amortized over 4 N-tiles);
// inner loop is pure b128 loads + 16 WMMA per 32-K step.
// mode 0: bf16 head-split [B,H,S,DH] (scaled);  mode 2: transposed [B,H,DH,S];
// mode 1: fp32 interleaved complex [M, N, 2] to d_out.
// ---------------------------------------------------------------------------
__global__ void __launch_bounds__(256)
cgemm_bf16(const __bf16* __restrict__ Xr, const __bf16* __restrict__ Xi,
           const __bf16* __restrict__ Br, const __bf16* __restrict__ Bi,
           const __bf16* __restrict__ Bn,
           __bf16* __restrict__ outR, __bf16* __restrict__ outI,
           float* __restrict__ outC, int mode, float scaleOut)
{
  const int K = CD;
  int wid  = (blockIdx.x << 3) + (threadIdx.x >> 5);
  int lane = threadIdx.x & 31;
  int half = lane >> 4, lq = lane & 15;
  int ng    = wid & 7;                        // 8 groups of 64 columns
  int mtile = wid >> 3;                       // 512 row tiles
  int m  = (mtile << 4) + lq;
  int n0 = ng << 6;

  v8f accR[4] = {}, accI[4] = {};
  const __bf16* xrp = Xr + m * K;
  const __bf16* xip = Xi + m * K;

  for (int kb = 0; kb < K; kb += 32) {
    int alo = kb + (half << 4);               // A elem j -> K=16h+j (contiguous 16)
    v16bf ar = concat8(ld8(xrp + alo), ld8(xrp + alo + 8));
    v16bf ai = concat8(ld8(xip + alo), ld8(xip + alo + 8));
    int bo = kb + (half << 4);                // B elem j -> K=16h+j (contiguous 16)
#pragma unroll
    for (int nt = 0; nt < 4; ++nt) {
      int n = n0 + (nt << 4) + lq;
      v16bf br = ld16(Br + n * K + bo);
      v16bf bi = ld16(Bi + n * K + bo);
      v16bf bn = ld16(Bn + n * K + bo);
      accR[nt] = wmma_bf16(ar, br, accR[nt]); // + xr*wr
      accR[nt] = wmma_bf16(ai, bn, accR[nt]); // - xi*wi
      accI[nt] = wmma_bf16(ar, bi, accI[nt]); // + xr*wi
      accI[nt] = wmma_bf16(ai, br, accI[nt]); // + xi*wr
    }
  }

#pragma unroll
  for (int nt = 0; nt < 4; ++nt) {
    int n = n0 + (nt << 4) + lq;
#pragma unroll
    for (int r = 0; r < 8; ++r) {
      int mrow = (mtile << 4) + (half << 3) + r;
      int bb = mrow >> 10, s = mrow & (CS - 1);
      int hh = n >> 6, d = n & (CDH - 1);
      if (mode == 0) {
        int idx = ((bb * CH + hh) * CS + s) * CDH + d;
        outR[idx] = f2bf(accR[nt][r] * scaleOut);
        outI[idx] = f2bf(accI[nt][r] * scaleOut);
      } else if (mode == 2) {                 // V stored transposed [B,H,DH,S]
        int idx = ((bb * CH + hh) * CDH + d) * CS + s;
        outR[idx] = f2bf(accR[nt][r]);
        outI[idx] = f2bf(accI[nt][r]);
      } else {
        float2* o = (float2*)outC;
        o[mrow * CD + n] = make_float2(accR[nt][r], accI[nt][r]);
      }
    }
  }
}

// ---------------------------------------------------------------------------
// Flash-style complex attention. Block = one (b,h) x 128 queries (8 waves x 16).
// 32-key K/V tiles staged in LDS by the Tensor Data Mover (double-buffered,
// wave 0 issues, s_wait_tensorcnt + workgroup barrier). S^T = K*Q^T trick keeps
// softmax per-lane and the exp'ed P tile directly in A-fragment layout.
// Outputs bf16 planes [8192, 512] (r/i) for the O-projection GEMM.
// ---------------------------------------------------------------------------
__global__ void __launch_bounds__(256)
cattn(const __bf16* __restrict__ qr, const __bf16* __restrict__ qi,
      const __bf16* __restrict__ kr, const __bf16* __restrict__ ki,
      const __bf16* __restrict__ vrT, const __bf16* __restrict__ viT,
      __bf16* __restrict__ oxr, __bf16* __restrict__ oxi)
{
  __shared__ __attribute__((aligned(32))) __bf16 skr[2][32 * 64]; // [key][d]
  __shared__ __attribute__((aligned(32))) __bf16 ski[2][32 * 64];
  __shared__ __attribute__((aligned(32))) __bf16 svr[2][64 * 32]; // [d][key]
  __shared__ __attribute__((aligned(32))) __bf16 svi[2][64 * 32];

  int warp = threadIdx.x >> 5;
  int lane = threadIdx.x & 31;
  int half = lane >> 4, lq = lane & 15;
  int bh = blockIdx.x >> 3;                  // 64 (b,h) pairs, 8 blocks each
  int qtile = ((blockIdx.x & 7) << 3) + warp;
  int b = bh >> 3, h = bh & 7;
  const int base = bh * (CS * CDH);

  // Q as B-fragments (lane = query column); 1/sqrt(DH) folded in by projection
  v16bf qrB[2], qiB[2], qiBn[2];
#pragma unroll
  for (int c = 0; c < 2; ++c) {
    const __bf16* pr = qr + base + ((qtile << 4) + lq) * CDH + (c << 5) + (half << 4);
    const __bf16* pi = qi + base + ((qtile << 4) + lq) * CDH + (c << 5) + (half << 4);
    qrB[c] = ld16(pr);
    v16bf xi = ld16(pi);
    qiB[c] = xi;
#pragma unroll
    for (int j = 0; j < 16; ++j) {
      unsigned short u = (unsigned short)(__builtin_bit_cast(unsigned short, xi[j]) ^ 0x8000u);
      qiBn[c][j] = __builtin_bit_cast(__bf16, u);
    }
  }

  v8f Urr[4] = {}, Uri[4] = {}, Uir[4] = {}, Uii[4] = {};
  float mR = -1e30f, lR = 0.f, mI = -1e30f, lI = 0.f;
  const int NKB = CS / 32;                   // 32 key blocks

  if (warp == 0) {
    tdm_load_2d(ldsOff(skr[0]), kr  + base, CDH, CS, CDH, CDH, 32);
    tdm_load_2d(ldsOff(ski[0]), ki  + base, CDH, CS, CDH, CDH, 32);
    tdm_load_2d(ldsOff(svr[0]), vrT + base, CS, CDH, CS, 32, CDH);
    tdm_load_2d(ldsOff(svi[0]), viT + base, CS, CDH, CS, 32, CDH);
  }

  for (int it = 0; it < NKB; ++it) {
    int buf = it & 1;
    if (warp == 0) {
      if (it + 1 < NKB) {
        int kb = (it + 1) * 32;
        tdm_load_2d(ldsOff(skr[buf ^ 1]), kr  + base + kb * CDH, CDH, CS, CDH, CDH, 32);
        tdm_load_2d(ldsOff(ski[buf ^ 1]), ki  + base + kb * CDH, CDH, CS, CDH, CDH, 32);
        tdm_load_2d(ldsOff(svr[buf ^ 1]), vrT + base + kb,       CS, CDH, CS, 32, CDH);
        tdm_load_2d(ldsOff(svi[buf ^ 1]), viT + base + kb,       CS, CDH, CS, 32, CDH);
        __builtin_amdgcn_s_wait_tensorcnt(4);   // current buffer's 4 loads done
      } else {
        __builtin_amdgcn_s_wait_tensorcnt(0);
      }
    }
    __syncthreads();                         // LDS tiles visible to all waves

    // K rows as A-fragments from LDS: 2 key sub-tiles x 2 d-chunks
    v16bf krA[2][2], kiA[2][2];
#pragma unroll
    for (int t = 0; t < 2; ++t)
#pragma unroll
      for (int c = 0; c < 2; ++c) {
        int row = ((t << 4) + lq) * CDH + (c << 5) + (half << 4);
        krA[t][c] = concat8(ld8(&skr[buf][row]), ld8(&skr[buf][row + 8]));
        kiA[t][c] = concat8(ld8(&ski[buf][row]), ld8(&ski[buf][row + 8]));
      }

    // S^T tiles: M = key, N = query (lane holds one query, 8 keys in regs)
    v8f stR[2], stI[2];
#pragma unroll
    for (int t = 0; t < 2; ++t) {
      v8f z = {};
      z = wmma_bf16(krA[t][0], qrB[0],  z);
      z = wmma_bf16(krA[t][1], qrB[1],  z);
      z = wmma_bf16(kiA[t][0], qiBn[0], z);  // s_r = qr*kr - qi*ki
      z = wmma_bf16(kiA[t][1], qiBn[1], z);
      stR[t] = z;
      v8f w = {};
      w = wmma_bf16(krA[t][0], qiB[0], w);
      w = wmma_bf16(krA[t][1], qiB[1], w);
      w = wmma_bf16(kiA[t][0], qrB[0], w);   // s_i = qr*ki + qi*kr
      w = wmma_bf16(kiA[t][1], qrB[1], w);
      stI[t] = w;
    }

    // Online softmax over keys (per query = lq, replicated across halves)
    float tmaxR = -1e30f, tmaxI = -1e30f;
#pragma unroll
    for (int t = 0; t < 2; ++t)
#pragma unroll
      for (int r = 0; r < 8; ++r) {
        tmaxR = fmaxf(tmaxR, stR[t][r]);
        tmaxI = fmaxf(tmaxI, stI[t][r]);
      }
    tmaxR = fmaxf(tmaxR, __shfl_xor(tmaxR, 16, 32));
    tmaxI = fmaxf(tmaxI, __shfl_xor(tmaxI, 16, 32));
    float mRn = fmaxf(mR, tmaxR), mIn = fmaxf(mI, tmaxI);
    float facR = __expf(mR - mRn), facI = __expf(mI - mIn);

    v16bf pR, pI;
    float sR = 0.f, sI = 0.f;
#pragma unroll
    for (int t = 0; t < 2; ++t)
#pragma unroll
      for (int r = 0; r < 8; ++r) {
        float eR = __expf(stR[t][r] - mRn);
        float eI = __expf(stI[t][r] - mIn);
        sR += eR; sI += eI;
        pR[t * 8 + r] = f2bf(eR);            // A-frag layout: own-lane regs only
        pI[t * 8 + r] = f2bf(eI);
      }
    sR += __shfl_xor(sR, 16, 32);
    sI += __shfl_xor(sI, 16, 32);
    lR = lR * facR + sR;
    lI = lI * facI + sI;
    mR = mRn; mI = mIn;

    // Rescale accumulators (accumulator row q = 8*half + r)
    float frs[8], fis[8];
#pragma unroll
    for (int r = 0; r < 8; ++r) {
      int srcq = (half << 3) + r;
      frs[r] = __shfl(facR, srcq, 32);
      fis[r] = __shfl(facI, srcq, 32);
    }
#pragma unroll
    for (int dt = 0; dt < 4; ++dt)
#pragma unroll
      for (int r = 0; r < 8; ++r) {
        Urr[dt][r] *= frs[r];
        Uri[dt][r] *= frs[r];
        Uir[dt][r] *= fis[r];
        Uii[dt][r] *= fis[r];
      }

    // V as B-fragments from transposed LDS tile, accumulate four products
#pragma unroll
    for (int dt = 0; dt < 4; ++dt) {
      int row = ((dt << 4) + lq) * 32 + (half << 3);
      v16bf vrB = concat8(ld8(&svr[buf][row]), ld8(&svr[buf][row + 16]));
      v16bf viB = concat8(ld8(&svi[buf][row]), ld8(&svi[buf][row + 16]));
      Urr[dt] = wmma_bf16(pR, vrB, Urr[dt]);
      Uri[dt] = wmma_bf16(pR, viB, Uri[dt]);
      Uir[dt] = wmma_bf16(pI, vrB, Uir[dt]);
      Uii[dt] = wmma_bf16(pI, viB, Uii[dt]);
    }
    __syncthreads();                         // reads done before next TDM write
  }

  // Finalize: o_r = Arr*Vr/lR - Aii*Vi/lI ; o_i = Ar*Vi/lR + Ai*Vr/lI
  float invR = 1.f / lR, invI = 1.f / lI;
  float lrr[8], lii[8];
#pragma unroll
  for (int r = 0; r < 8; ++r) {
    int srcq = (half << 3) + r;
    lrr[r] = __shfl(invR, srcq, 32);
    lii[r] = __shfl(invI, srcq, 32);
  }
#pragma unroll
  for (int dt = 0; dt < 4; ++dt)
#pragma unroll
    for (int r = 0; r < 8; ++r) {
      int q = (qtile << 4) + (half << 3) + r;
      int m = b * CS + q;
      int dcol = h * CDH + (dt << 4) + lq;
      oxr[m * CD + dcol] = f2bf(Urr[dt][r] * lrr[r] - Uii[dt][r] * lii[r]);
      oxi[m * CD + dcol] = f2bf(Uri[dt][r] * lrr[r] + Uir[dt][r] * lii[r]);
    }
}

// ---------------------------------------------------------------------------
extern "C" void kernel_launch(void* const* d_in, const int* in_sizes, int n_in,
                              void* d_out, int out_size, void* d_ws, size_t ws_size,
                              hipStream_t stream) {
  const float* queries = (const float*)d_in[0];
  const float* keys    = (const float*)d_in[1];
  const float* values  = (const float*)d_in[2];
  const float* wq_r = (const float*)d_in[3];
  const float* wq_i = (const float*)d_in[4];
  const float* wk_r = (const float*)d_in[5];
  const float* wk_i = (const float*)d_in[6];
  const float* wv_r = (const float*)d_in[7];
  const float* wv_i = (const float*)d_in[8];
  const float* wo_r = (const float*)d_in[9];
  const float* wo_i = (const float*)d_in[10];

  const size_t PLANE = (size_t)CB * CS * CD;           // 4,194,304 elements
  const size_t WEL   = (size_t)CD * CD;                // 262,144 elements
  char* ws = (char*)d_ws;
  __bf16* sxr = (__bf16*)(ws + 0 * PLANE * 2);         // staging / attn out (r)
  __bf16* sxi = (__bf16*)(ws + 1 * PLANE * 2);         // staging / attn out (i)
  __bf16* qrP = (__bf16*)(ws + 2 * PLANE * 2);
  __bf16* qiP = (__bf16*)(ws + 3 * PLANE * 2);
  __bf16* krP = (__bf16*)(ws + 4 * PLANE * 2);
  __bf16* kiP = (__bf16*)(ws + 5 * PLANE * 2);
  __bf16* vrP = (__bf16*)(ws + 6 * PLANE * 2);
  __bf16* viP = (__bf16*)(ws + 7 * PLANE * 2);
  __bf16* wbr = (__bf16*)(ws + 8 * PLANE * 2);
  __bf16* wbi = wbr + WEL;
  __bf16* wbn = wbi + WEL;

  dim3 blk(256);
  const int GX = (int)(PLANE / 256);                   // 16384
  const int GW = (int)(WEL / 256);                     // 1024

  // Q projection (scale 1/sqrt(64) folded into q planes)
  conv_x<<<GX, blk, 0, stream>>>((const float2*)queries, sxr, sxi);
  conv_w<<<GW, blk, 0, stream>>>(wq_r, wq_i, wbr, wbi, wbn);
  cgemm_bf16<<<512, blk, 0, stream>>>(sxr, sxi, wbr, wbi, wbn, qrP, qiP, nullptr, 0, 0.125f);
  // K projection
  conv_x<<<GX, blk, 0, stream>>>((const float2*)keys, sxr, sxi);
  conv_w<<<GW, blk, 0, stream>>>(wk_r, wk_i, wbr, wbi, wbn);
  cgemm_bf16<<<512, blk, 0, stream>>>(sxr, sxi, wbr, wbi, wbn, krP, kiP, nullptr, 0, 1.0f);
  // V projection (stored transposed [B,H,DH,S] for TDM/B-fragment contiguity)
  conv_x<<<GX, blk, 0, stream>>>((const float2*)values, sxr, sxi);
  conv_w<<<GW, blk, 0, stream>>>(wv_r, wv_i, wbr, wbi, wbn);
  cgemm_bf16<<<512, blk, 0, stream>>>(sxr, sxi, wbr, wbi, wbn, vrP, viP, nullptr, 2, 1.0f);
  // Attention (writes bf16 planes into the staging buffers)
  cattn<<<512, blk, 0, stream>>>(qrP, qiP, krP, kiP, vrP, viP, sxr, sxi);
  // Output projection -> fp32 interleaved complex d_out
  conv_w<<<GW, blk, 0, stream>>>(wo_r, wo_i, wbr, wbi, wbn);
  cgemm_bf16<<<512, blk, 0, stream>>>(sxr, sxi, wbr, wbi, wbn, nullptr, nullptr,
                                      (float*)d_out, 1, 1.0f);
}
